// EnergySharingNetwork_54039278518761
// MI455X (gfx1250) — matrix operand order repaired and checked by reading, not compiled, not run on hardware
//
#include <hip/hip_runtime.h>
#include <hip/hip_bf16.h>

#define N_NODES 50000
#define N_EDGES 1200000

typedef __bf16 bf16_t;
typedef __attribute__((ext_vector_type(16))) __bf16 v16bf;
typedef __attribute__((ext_vector_type(8)))  __bf16 v8bf;
typedef __attribute__((ext_vector_type(8)))  float   v8f;

static __device__ __forceinline__ v8f wmma_bf16(v16bf a, v16bf b, v8f c) {
  return __builtin_amdgcn_wmma_f32_16x16x32_bf16(false, a, false, b, (short)0, c, false, false);
}

// A-fragment per ISA 16-bit A 16x32 layout:
// lane<16 : row=lane,    K = {k0..k0+7} in v0..3, {k0+16..k0+23} in v4..7 (k0 = 0)
// lane>=16: row=lane-16, K = {8..15}, {24..31}  -> caller passes k0 = half*8
static __device__ __forceinline__ v16bf load_afrag(const bf16_t* p, int k0) {
  v8bf lo = *(const v8bf*)(p + k0);
  v8bf hi = *(const v8bf*)(p + k0 + 16);
  return __builtin_shufflevector(lo, hi, 0,1,2,3,4,5,6,7,8,9,10,11,12,13,14,15);
}

// ---------------- prep: transpose + convert weights to bf16 W^T[n][k] ----------------
__global__ void prep_weights(const float* __restrict__ Wn2, const float* __restrict__ Wm1,
                             const float* __restrict__ Wm2, const float* __restrict__ Wu1,
                             const float* __restrict__ Wu2,
                             bf16_t* __restrict__ wn2t, bf16_t* __restrict__ wm1t,
                             bf16_t* __restrict__ wm2t, bf16_t* __restrict__ wu1t,
                             bf16_t* __restrict__ wu2t) {
  const int total = (64 + 192 + 64 + 128 + 64) * 64; // 32768
  for (int i = blockIdx.x * blockDim.x + threadIdx.x; i < total; i += gridDim.x * blockDim.x) {
    int rem = i;
    const float* W; bf16_t* WT; int K;
    if (rem < 64 * 64)                 { W = Wn2; WT = wn2t; K = 64; }
    else if ((rem -= 64 * 64) < 192*64){ W = Wm1; WT = wm1t; K = 192; }
    else if ((rem -= 192 * 64) < 64*64){ W = Wm2; WT = wm2t; K = 64; }
    else if ((rem -= 64 * 64) < 128*64){ W = Wu1; WT = wu1t; K = 128; }
    else { rem -= 128 * 64;              W = Wu2; WT = wu2t; K = 64; }
    int k = rem / 64, n = rem % 64;
    WT[n * K + k] = (bf16_t)W[k * 64 + n];
  }
}

__global__ void zero_f32(float* __restrict__ p, int n) {
  for (int i = blockIdx.x * blockDim.x + threadIdx.x; i < n; i += gridDim.x * blockDim.x)
    p[i] = 0.f;
}

// ---------------- kernel 1: node encoder h = relu(x@Wn1+b)@Wn2+b (bf16 out) ----------------
__global__ __launch_bounds__(256) void node_encoder_kernel(
    const float* __restrict__ x, const float* __restrict__ Wn1, const float* __restrict__ bn1,
    const float* __restrict__ bn2, const bf16_t* __restrict__ wn2t, bf16_t* __restrict__ hb) {
  __shared__ float sWn1[5 * 64];
  __shared__ float sB1[64], sB2[64];
  __shared__ __attribute__((aligned(32))) bf16_t sW2[64 * 64];
  __shared__ __attribute__((aligned(32))) bf16_t sA[8][16 * 64];

  const int tid = threadIdx.x;
  for (int i = tid; i < 5 * 64; i += 256) sWn1[i] = Wn1[i];
  for (int i = tid; i < 64; i += 256) { sB1[i] = bn1[i]; sB2[i] = bn2[i]; }
  { const uint32_t* s = (const uint32_t*)wn2t; uint32_t* d = (uint32_t*)sW2;
    for (int i = tid; i < 64 * 64 / 2; i += 256) d[i] = s[i]; }
  __syncthreads();

  const int wid = tid >> 5, lane = tid & 31, half = lane >> 4, row = lane & 15;
  bf16_t* myA = sA[wid];
  const int nTiles = N_NODES / 16, totalWaves = gridDim.x * 8;
  for (int tile = blockIdx.x * 8 + wid; tile < nTiles; tile += totalWaves) {
    const int base = tile * 16, node = base + row;
    float xr[5];
    #pragma unroll
    for (int k = 0; k < 5; ++k) xr[k] = x[node * 5 + k];
    #pragma unroll
    for (int cc = 0; cc < 32; ++cc) {
      int c = half * 32 + cc;
      float t = sB1[c];
      #pragma unroll
      for (int k = 0; k < 5; ++k) t += xr[k] * sWn1[k * 64 + c];
      myA[row * 64 + c] = (bf16_t)(t > 0.f ? t : 0.f);
    }
    v16bf a0 = load_afrag(myA, half * 8);
    v16bf a1 = load_afrag(myA, 32 + half * 8);
    #pragma unroll
    for (int nt = 0; nt < 4; ++nt) {
      v8f acc = {0, 0, 0, 0, 0, 0, 0, 0};
      v16bf b0 = load_afrag(&sW2[(nt * 16 + row) * 64], half * 8);
      v16bf b1 = load_afrag(&sW2[(nt * 16 + row) * 64 + 32], half * 8);
      acc = wmma_bf16(a0, b0, acc);
      acc = wmma_bf16(a1, b1, acc);
      float bias = sB2[nt * 16 + row];
      #pragma unroll
      for (int i = 0; i < 8; ++i)
        hb[(size_t)(base + i + 8 * half) * 64 + nt * 16 + row] = (bf16_t)(acc[i] + bias);
    }
  }
}

// ---------------- kernel 2: messages + atomic mean-aggregation ----------------
__global__ __launch_bounds__(256) void message_kernel(
    const int* __restrict__ ei, const float* __restrict__ edge_attr,
    const float* __restrict__ We, const float* __restrict__ be,
    const float* __restrict__ bm1, const float* __restrict__ bm2,
    const bf16_t* __restrict__ wm1t, const bf16_t* __restrict__ wm2t,
    const bf16_t* __restrict__ hb, float* __restrict__ msum, float* __restrict__ cnt) {
  __shared__ __attribute__((aligned(32))) bf16_t sW1[64 * 192];
  __shared__ __attribute__((aligned(32))) bf16_t sW2[64 * 64];
  __shared__ float sWe[128], sBe[64], sB1[64], sB2[64];
  __shared__ __attribute__((aligned(32))) bf16_t sA[8][16 * 64];
  __shared__ int sDst[8][16];

  const int tid = threadIdx.x;
  { const uint32_t* s = (const uint32_t*)wm1t; uint32_t* d = (uint32_t*)sW1;
    for (int i = tid; i < 64 * 192 / 2; i += 256) d[i] = s[i]; }
  { const uint32_t* s = (const uint32_t*)wm2t; uint32_t* d = (uint32_t*)sW2;
    for (int i = tid; i < 64 * 64 / 2; i += 256) d[i] = s[i]; }
  for (int i = tid; i < 128; i += 256) sWe[i] = We[i];
  for (int i = tid; i < 64; i += 256) { sBe[i] = be[i]; sB1[i] = bm1[i]; sB2[i] = bm2[i]; }
  __syncthreads();

  const int wid = tid >> 5, lane = tid & 31, half = lane >> 4, row = lane & 15;
  bf16_t* myA = sA[wid];
  int* myDst = sDst[wid];
  const int nTiles = N_EDGES / 16, totalWaves = gridDim.x * 8;
  for (int tile = blockIdx.x * 8 + wid; tile < nTiles; tile += totalWaves) {
    const int eid = tile * 16 + row;
    const int src = ei[eid];
    const int dst = ei[N_EDGES + eid];
    if (half == 0) myDst[row] = dst;

    const bf16_t* hD = hb + (size_t)dst * 64;
    const bf16_t* hS = hb + (size_t)src * 64;
    const int k0 = half * 8;
    v16bf a[6];
    a[0] = load_afrag(hD, k0);       // m_in cols 0..63  = h[dst]
    a[1] = load_afrag(hD, 32 + k0);
    a[2] = load_afrag(hS, k0);       // cols 64..127 = h[src]
    a[3] = load_afrag(hS, 32 + k0);
    const float ea0 = edge_attr[2 * eid], ea1 = edge_attr[2 * eid + 1];
    #pragma unroll
    for (int t = 0; t < 2; ++t) {    // cols 128..191 = e = ea@We + be (computed on the fly)
      v16bf v;
      #pragma unroll
      for (int j = 0; j < 16; ++j) {
        int c = t * 32 + k0 + (j < 8 ? j : j + 8);
        v[j] = (bf16_t)(ea0 * sWe[c] + ea1 * sWe[64 + c] + sBe[c]);
      }
      a[4 + t] = v;
    }
    // layer 1: [16,192]@[192,64], ReLU -> LDS tile
    #pragma unroll
    for (int nt = 0; nt < 4; ++nt) {
      v8f acc = {0, 0, 0, 0, 0, 0, 0, 0};
      #pragma unroll
      for (int kt = 0; kt < 6; ++kt) {
        v16bf b = load_afrag(&sW1[(nt * 16 + row) * 192 + kt * 32], half * 8);
        acc = wmma_bf16(a[kt], b, acc);
      }
      float bias = sB1[nt * 16 + row];
      #pragma unroll
      for (int i = 0; i < 8; ++i) {
        float v = acc[i] + bias;
        myA[(i + 8 * half) * 64 + nt * 16 + row] = (bf16_t)(v > 0.f ? v : 0.f);
      }
    }
    // layer 2: [16,64]@[64,64] + scatter-add
    v16bf a20 = load_afrag(myA, half * 8);
    v16bf a21 = load_afrag(myA, 32 + half * 8);
    if (half == 0) unsafeAtomicAdd(&cnt[dst], 1.0f);
    #pragma unroll
    for (int nt = 0; nt < 4; ++nt) {
      v8f acc = {0, 0, 0, 0, 0, 0, 0, 0};
      v16bf b0 = load_afrag(&sW2[(nt * 16 + row) * 64], half * 8);
      v16bf b1 = load_afrag(&sW2[(nt * 16 + row) * 64 + 32], half * 8);
      acc = wmma_bf16(a20, b0, acc);
      acc = wmma_bf16(a21, b1, acc);
      float bias = sB2[nt * 16 + row];
      #pragma unroll
      for (int i = 0; i < 8; ++i)
        unsafeAtomicAdd(&msum[(size_t)myDst[i + 8 * half] * 64 + nt * 16 + row], acc[i] + bias);
    }
  }
}

// ---------------- kernel 3: out = relu([h|agg]@Wu1+b)@Wu2+b ----------------
__global__ __launch_bounds__(256) void update_kernel(
    const bf16_t* __restrict__ hb, const float* __restrict__ msum, const float* __restrict__ cnt,
    const float* __restrict__ bu1, const float* __restrict__ bu2,
    const bf16_t* __restrict__ wu1t, const bf16_t* __restrict__ wu2t, float* __restrict__ out) {
  __shared__ __attribute__((aligned(32))) bf16_t sW1[64 * 128];
  __shared__ __attribute__((aligned(32))) bf16_t sW2[64 * 64];
  __shared__ float sB1[64], sB2[64];
  __shared__ __attribute__((aligned(32))) bf16_t sA[8][16 * 64];

  const int tid = threadIdx.x;
  { const uint32_t* s = (const uint32_t*)wu1t; uint32_t* d = (uint32_t*)sW1;
    for (int i = tid; i < 64 * 128 / 2; i += 256) d[i] = s[i]; }
  { const uint32_t* s = (const uint32_t*)wu2t; uint32_t* d = (uint32_t*)sW2;
    for (int i = tid; i < 64 * 64 / 2; i += 256) d[i] = s[i]; }
  for (int i = tid; i < 64; i += 256) { sB1[i] = bu1[i]; sB2[i] = bu2[i]; }
  __syncthreads();

  const int wid = tid >> 5, lane = tid & 31, half = lane >> 4, row = lane & 15;
  bf16_t* myA = sA[wid];
  const int nTiles = N_NODES / 16, totalWaves = gridDim.x * 8;
  for (int tile = blockIdx.x * 8 + wid; tile < nTiles; tile += totalWaves) {
    const int base = tile * 16, node = base + row;
    const int k0 = half * 8;
    v16bf a[4];
    const bf16_t* hR = hb + (size_t)node * 64;
    a[0] = load_afrag(hR, k0);        // cols 0..63 = h
    a[1] = load_afrag(hR, 32 + k0);
    const float inv = 1.f / fmaxf(cnt[node], 1.f);
    const float* ms = msum + (size_t)node * 64;
    #pragma unroll
    for (int t = 0; t < 2; ++t) {     // cols 64..127 = agg = msum/cnt
      v16bf v;
      #pragma unroll
      for (int j = 0; j < 16; ++j) {
        int c = t * 32 + k0 + (j < 8 ? j : j + 8);
        v[j] = (bf16_t)(ms[c] * inv);
      }
      a[2 + t] = v;
    }
    #pragma unroll
    for (int nt = 0; nt < 4; ++nt) {
      v8f acc = {0, 0, 0, 0, 0, 0, 0, 0};
      #pragma unroll
      for (int kt = 0; kt < 4; ++kt) {
        v16bf b = load_afrag(&sW1[(nt * 16 + row) * 128 + kt * 32], half * 8);
        acc = wmma_bf16(a[kt], b, acc);
      }
      float bias = sB1[nt * 16 + row];
      #pragma unroll
      for (int i = 0; i < 8; ++i) {
        float v = acc[i] + bias;
        myA[(i + 8 * half) * 64 + nt * 16 + row] = (bf16_t)(v > 0.f ? v : 0.f);
      }
    }
    v16bf a20 = load_afrag(myA, half * 8);
    v16bf a21 = load_afrag(myA, 32 + half * 8);
    #pragma unroll
    for (int nt = 0; nt < 4; ++nt) {
      v8f acc = {0, 0, 0, 0, 0, 0, 0, 0};
      v16bf b0 = load_afrag(&sW2[(nt * 16 + row) * 64], half * 8);
      v16bf b1 = load_afrag(&sW2[(nt * 16 + row) * 64 + 32], half * 8);
      acc = wmma_bf16(a20, b0, acc);
      acc = wmma_bf16(a21, b1, acc);
      float bias = sB2[nt * 16 + row];
      #pragma unroll
      for (int i = 0; i < 8; ++i)
        out[(size_t)(base + i + 8 * half) * 64 + nt * 16 + row] = acc[i] + bias;
    }
  }
}

extern "C" void kernel_launch(void* const* d_in, const int* in_sizes, int n_in,
                              void* d_out, int out_size, void* d_ws, size_t ws_size,
                              hipStream_t stream) {
  const float* x         = (const float*)d_in[0];
  const float* edge_attr = (const float*)d_in[1];
  const int*   edge_idx  = (const int*)d_in[2];
  const float* Wn1 = (const float*)d_in[3];  const float* bn1 = (const float*)d_in[4];
  const float* Wn2 = (const float*)d_in[5];  const float* bn2 = (const float*)d_in[6];
  const float* We  = (const float*)d_in[7];  const float* be  = (const float*)d_in[8];
  const float* Wm1 = (const float*)d_in[9];  const float* bm1 = (const float*)d_in[10];
  const float* Wm2 = (const float*)d_in[11]; const float* bm2 = (const float*)d_in[12];
  const float* Wu1 = (const float*)d_in[13]; const float* bu1 = (const float*)d_in[14];
  const float* Wu2 = (const float*)d_in[15]; const float* bu2 = (const float*)d_in[16];

  char* ws = (char*)d_ws;
  bf16_t* wn2t = (bf16_t*)(ws + 0);            //  8 KB
  bf16_t* wm1t = (bf16_t*)(ws + 8192);         // 24 KB
  bf16_t* wm2t = (bf16_t*)(ws + 32768);        //  8 KB
  bf16_t* wu1t = (bf16_t*)(ws + 40960);        // 16 KB
  bf16_t* wu2t = (bf16_t*)(ws + 57344);        //  8 KB
  bf16_t* hb   = (bf16_t*)(ws + 65536);        // N*64 bf16 = 6.4 MB
  float*  msum = (float*)(ws + 6465536);       // N*64 f32  = 12.8 MB
  float*  cnt  = (float*)(ws + 19265536);      // N f32 (contiguous after msum)

  prep_weights<<<64, 256, 0, stream>>>(Wn2, Wm1, Wm2, Wu1, Wu2, wn2t, wm1t, wm2t, wu1t, wu2t);
  zero_f32<<<2048, 256, 0, stream>>>(msum, N_NODES * 64 + N_NODES);
  node_encoder_kernel<<<391, 256, 0, stream>>>(x, Wn1, bn1, bn2, wn2t, hb);
  message_kernel<<<512, 256, 0, stream>>>(edge_idx, edge_attr, We, be, bm1, bm2,
                                          wm1t, wm2t, hb, msum, cnt);
  update_kernel<<<391, 256, 0, stream>>>(hb, msum, cnt, bu1, bu2, wu1t, wu2t, (float*)d_out);
}
